// AttentionSampling_75986561400936
// MI455X (gfx1250) — matrix operand order repaired and checked by reading, not compiled, not run on hardware
//
#include <hip/hip_runtime.h>
#include <hip/hip_bf16.h>

// ---------- types for CDNA5 WMMA ----------
typedef __attribute__((ext_vector_type(8)))  float  v8f;
typedef __attribute__((ext_vector_type(16))) __bf16 v16bf;
typedef __attribute__((ext_vector_type(8)))  __bf16 v8bf;

union V16U { v16bf v; v8bf h[2]; };

// ---------- block-wide reduction (wave32) ----------
__device__ __forceinline__ float blockReduceSum(float v) {
    __shared__ float red[8];
    #pragma unroll
    for (int o = 16; o > 0; o >>= 1) v += __shfl_down(v, o, 32);
    const int lane = threadIdx.x & 31;
    const int wave = threadIdx.x >> 5;
    if (lane == 0) red[wave] = v;
    __syncthreads();
    if (wave == 0) {
        float r = (lane < 8) ? red[lane] : 0.0f;
        #pragma unroll
        for (int o = 4; o > 0; o >>= 1) r += __shfl_down(r, o, 32);
        if (lane == 0) red[0] = r;
    }
    __syncthreads();
    float out = red[0];
    __syncthreads();
    return out;
}

// ---------- streaming fp32 -> bf16 convert (8 elems / thread) ----------
__global__ __launch_bounds__(256) void cvt_bf16(const float* __restrict__ in,
                                                __bf16* __restrict__ out, int n8) {
    int i = blockIdx.x * 256 + threadIdx.x;
    if (i < n8) {
        const float4 a = ((const float4*)in)[2 * i];
        const float4 b = ((const float4*)in)[2 * i + 1];
        v8bf o = { (__bf16)a.x, (__bf16)a.y, (__bf16)a.z, (__bf16)a.w,
                   (__bf16)b.x, (__bf16)b.y, (__bf16)b.z, (__bf16)b.w };
        ((v8bf*)out)[i] = o;
    }
}

// ---------- weight transpose+convert: W[K][N] fp32 -> Wt[N][K] bf16 ----------
__global__ __launch_bounds__(256) void wtrans512(const float* __restrict__ W,
                                                 __bf16* __restrict__ Wt) {
    int i = blockIdx.x * 256 + threadIdx.x;          // 512*512 elements
    int n = i >> 9;
    int k = i & 511;
    Wt[(size_t)n * 512 + k] = (__bf16)W[(size_t)k * 512 + n];
}

// ---------- bf16 WMMA GEMM with async-to-LDS pipeline ----------
// C[M,512] = act(A[M,512] @ W + bias); A bf16 [M][K], Wt bf16 [N][K].
// Block tile 128x128, 8 waves, wave tile 32x64.  4-deep LDS ring buffer filled
// with global_load_async_to_lds_b128 (ASYNCcnt), drained with s_wait_asynccnt.
template <int RELU, int OBF>
__global__ __launch_bounds__(256) void gemm_bf16_async(const __bf16* __restrict__ A,
                                                       const __bf16* __restrict__ Wt,
                                                       const float* __restrict__ bias,
                                                       float* __restrict__ Cf,
                                                       __bf16* __restrict__ Cb, int M) {
    constexpr int K = 512, N = 512;
    constexpr unsigned BUFB = 128u * 40u * 2u;       // bytes per LDS stage buffer
    __shared__ __bf16 As[4][128][40];                // 32 k + 8 pad
    __shared__ __bf16 Bs[4][128][40];

    const int t    = threadIdx.x;
    const int lane = t & 31;
    const int wave = t >> 5;
    const int wm   = wave >> 1;                      // 0..3
    const int wn   = wave & 1;                       // 0..1
    const int m0   = blockIdx.y * 128;
    const int n0   = blockIdx.x * 128;

    const unsigned long long Abase = (unsigned long long)A;
    const unsigned long long Bbase = (unsigned long long)Wt;

    // per-thread transfer slots: 512 b128 per tile / 256 thr = 2 each (A and B)
    unsigned ldsA0[2], ldsB0[2], voffA[2], voffB[2];
    #pragma unroll
    for (int j = 0; j < 2; ++j) {
        int e  = t + j * 256;
        int r  = e >> 2;                             // row/col of tile
        int kg = e & 3;                              // 16B group along k
        ldsA0[j] = (unsigned)(size_t)&As[0][r][kg * 8];
        ldsB0[j] = (unsigned)(size_t)&Bs[0][r][kg * 8];
        voffA[j] = (unsigned)(((m0 + r) * K + kg * 8) * 2);
        voffB[j] = (unsigned)(((n0 + r) * K + kg * 8) * 2);
    }

    v8f acc[2][4];
    #pragma unroll
    for (int mi = 0; mi < 2; ++mi)
        #pragma unroll
        for (int ni = 0; ni < 4; ++ni)
            acc[mi][ni] = (v8f){0.f, 0.f, 0.f, 0.f, 0.f, 0.f, 0.f, 0.f};

    const int frow = lane & 15;                      // row/col within 16-tile
    const int akb  = (lane >> 4) * 8;                // ISA 16-bit A frag K base
    const int bkb  = (lane >> 4) * 8;                // ISA 16-bit B frag K base

    // issue async loads for k-stage s into ring buffer s&3 (4 issues/thread)
    auto issue = [&](int s) {
        const unsigned kadd = (unsigned)(s * 64);    // 32 bf16 = 64 bytes per stage
        const unsigned boff = (unsigned)((s & 3) * BUFB);
        #pragma unroll
        for (int j = 0; j < 2; ++j) {
            asm volatile("global_load_async_to_lds_b128 %0, %1, %2"
                         :: "v"(ldsA0[j] + boff), "v"(voffA[j] + kadd), "s"(Abase)
                         : "memory");
            asm volatile("global_load_async_to_lds_b128 %0, %1, %2"
                         :: "v"(ldsB0[j] + boff), "v"(voffB[j] + kadd), "s"(Bbase)
                         : "memory");
        }
    };

    issue(0); issue(1); issue(2);                    // prime 3 stages
    asm volatile("s_wait_asynccnt 8" ::: "memory");  // stage 0 landed
    __syncthreads();

    for (int ks = 0; ks < 16; ++ks) {
        if (ks + 3 < 16) issue(ks + 3);
        const int p = ks & 3;

        V16U afrag[2], bfrag[4];
        #pragma unroll
        for (int mi = 0; mi < 2; ++mi) {
            const __bf16* pr = &As[p][wm * 32 + mi * 16 + frow][0];
            afrag[mi].h[0] = *(const v8bf*)(pr + akb);
            afrag[mi].h[1] = *(const v8bf*)(pr + akb + 16);
        }
        #pragma unroll
        for (int ni = 0; ni < 4; ++ni) {
            const __bf16* pc = &Bs[p][wn * 64 + ni * 16 + frow][0];
            bfrag[ni].h[0] = *(const v8bf*)(pc + bkb);
            bfrag[ni].h[1] = *(const v8bf*)(pc + bkb + 16);
        }
        #pragma unroll
        for (int mi = 0; mi < 2; ++mi)
            #pragma unroll
            for (int ni = 0; ni < 4; ++ni)
                acc[mi][ni] = __builtin_amdgcn_wmma_f32_16x16x32_bf16(
                    false, afrag[mi].v, false, bfrag[ni].v,
                    (short)0, acc[mi][ni], false, false);

        // before the publishing barrier: ensure stage ks+1 has landed.
        // per-wave in-order ASYNCcnt: allow only the 'rem' younger stages
        // (4 issues each) to remain outstanding.
        const int rem = 14 - ks;
        if (rem >= 2)      asm volatile("s_wait_asynccnt 8" ::: "memory");
        else if (rem == 1) asm volatile("s_wait_asynccnt 4" ::: "memory");
        else               asm volatile("s_wait_asynccnt 0" ::: "memory");
        __syncthreads();
    }

    // epilogue: bias + optional ReLU; fp32 or bf16 output
    const int mbase = m0 + wm * 32 + (lane >> 4) * 8;
    const int nbase = n0 + wn * 64 + (lane & 15);
    #pragma unroll
    for (int mi = 0; mi < 2; ++mi) {
        #pragma unroll
        for (int ni = 0; ni < 4; ++ni) {
            const int col  = nbase + ni * 16;
            const float bc = bias[col];
            #pragma unroll
            for (int r = 0; r < 8; ++r) {
                int row = mbase + mi * 16 + r;
                float val = acc[mi][ni][r] + bc;
                if (RELU) val = fmaxf(val, 0.0f);
                if (row < M) {
                    if (OBF) Cb[(size_t)row * N + col] = (__bf16)val;
                    else     Cf[(size_t)row * N + col] = val;
                }
            }
        }
    }
}

// ---------- attention downsampling + residual + LayerNorm1 ----------
__global__ __launch_bounds__(256) void attn_ln1(const float* __restrict__ q,
                                                const float* __restrict__ k,
                                                const float* __restrict__ v,
                                                const float* __restrict__ g,
                                                const float* __restrict__ beta,
                                                float* __restrict__ out1,
                                                __bf16* __restrict__ out1b) {
    const int t  = threadIdx.x;
    const int bs = blockIdx.x;               // b*2048 + s
    const int b  = bs >> 11;
    const int s  = bs & 2047;
    const float* qr = q + (size_t)bs * 512;
    const size_t kvbase = ((size_t)b * 8192 + (size_t)4 * s) * 512;

    const float q0 = qr[t], q1 = qr[t + 256];

    float w[4];
    #pragma unroll
    for (int f = 0; f < 4; ++f) {
        const float* kr = k + kvbase + (size_t)f * 512;
        w[f] = blockReduceSum(q0 * kr[t] + q1 * kr[t + 256]);
    }

    float x0 = q0, x1 = q1;
    #pragma unroll
    for (int f = 0; f < 4; ++f) {
        const float* vr = v + kvbase + (size_t)f * 512;
        x0 += w[f] * vr[t];
        x1 += w[f] * vr[t + 256];
    }

    const float mu  = blockReduceSum(x0 + x1) * (1.0f / 512.0f);
    const float d0  = x0 - mu, d1 = x1 - mu;
    const float var = blockReduceSum(d0 * d0 + d1 * d1) * (1.0f / 512.0f);
    const float rs  = rsqrtf(var + 1e-5f);

    const float y0 = d0 * rs * g[t] + beta[t];
    const float y1 = d1 * rs * g[t + 256] + beta[t + 256];
    out1 [(size_t)bs * 512 + t]        = y0;
    out1 [(size_t)bs * 512 + t + 256]  = y1;
    out1b[(size_t)bs * 512 + t]        = (__bf16)y0;
    out1b[(size_t)bs * 512 + t + 256]  = (__bf16)y1;
}

// ---------- residual + LayerNorm2 -> d_out ----------
__global__ __launch_bounds__(256) void add_ln2(const float* __restrict__ a,
                                               const float* __restrict__ f,
                                               const float* __restrict__ g,
                                               const float* __restrict__ beta,
                                               float* __restrict__ out) {
    const int t = threadIdx.x;
    const size_t base = (size_t)blockIdx.x * 512;
    float x0 = a[base + t] + f[base + t];
    float x1 = a[base + t + 256] + f[base + t + 256];

    const float mu  = blockReduceSum(x0 + x1) * (1.0f / 512.0f);
    const float d0  = x0 - mu, d1 = x1 - mu;
    const float var = blockReduceSum(d0 * d0 + d1 * d1) * (1.0f / 512.0f);
    const float rs  = rsqrtf(var + 1e-5f);

    out[base + t]       = d0 * rs * g[t] + beta[t];
    out[base + t + 256] = d1 * rs * g[t + 256] + beta[t + 256];
}

extern "C" void kernel_launch(void* const* d_in, const int* in_sizes, int n_in,
                              void* d_out, int out_size, void* d_ws, size_t ws_size,
                              hipStream_t stream) {
    (void)in_sizes; (void)n_in; (void)out_size; (void)ws_size;

    const float* query = (const float*)d_in[0];
    const float* key   = (const float*)d_in[1];
    const float* value = (const float*)d_in[2];
    const float* w_q   = (const float*)d_in[3];
    const float* b_q   = (const float*)d_in[4];
    const float* w_k   = (const float*)d_in[5];
    const float* b_k   = (const float*)d_in[6];
    const float* w_v   = (const float*)d_in[7];
    const float* b_v   = (const float*)d_in[8];
    const float* ln1_g = (const float*)d_in[9];
    const float* ln1_b = (const float*)d_in[10];
    const float* ln2_g = (const float*)d_in[11];
    const float* ln2_b = (const float*)d_in[12];
    const float* fw1   = (const float*)d_in[13];
    const float* fb1   = (const float*)d_in[14];
    const float* fw2   = (const float*)d_in[15];
    const float* fb2   = (const float*)d_in[16];

    char* p = (char*)d_ws;
    const size_t MQ = (size_t)8192  * 512;   // q-side rows*D
    const size_t MK = (size_t)32768 * 512;   // kv-side rows*D
    float*  qb  = (float*)p;  p += MQ * 4;   // 16 MB
    float*  kb  = (float*)p;  p += MK * 4;   // 64 MB
    float*  vb  = (float*)p;  p += MK * 4;   // 64 MB
    float*  o1  = (float*)p;  p += MQ * 4;   // 16 MB
    __bf16* o1b = (__bf16*)p; p += MQ * 2;   //  8 MB
    __bf16* qbf = (__bf16*)p; p += MQ * 2;   //  8 MB
    __bf16* kbf = (__bf16*)p; p += MK * 2;   // 32 MB
    __bf16* vbf = (__bf16*)p; p += MK * 2;   // 32 MB
    __bf16* wt  = (__bf16*)p; p += (size_t)5 * 512 * 512 * 2;  // 2.5 MB
    __bf16* wt_q = wt;
    __bf16* wt_k = wt + (size_t)1 * 512 * 512;
    __bf16* wt_v = wt + (size_t)2 * 512 * 512;
    __bf16* wt_1 = wt + (size_t)3 * 512 * 512;
    __bf16* wt_2 = wt + (size_t)4 * 512 * 512;
    __bf16* hidb   = qbf;   // qbf dead after Q projection
    float*  ffnout = vb;    // vb dead after attn_ln1
    float*  outp   = (float*)d_out;

    // 1) weights -> bf16 [N][K]; activations -> bf16
    wtrans512<<<1024, 256, 0, stream>>>(w_q, wt_q);
    wtrans512<<<1024, 256, 0, stream>>>(w_k, wt_k);
    wtrans512<<<1024, 256, 0, stream>>>(w_v, wt_v);
    wtrans512<<<1024, 256, 0, stream>>>(fw1, wt_1);
    wtrans512<<<1024, 256, 0, stream>>>(fw2, wt_2);
    cvt_bf16<<<2048, 256, 0, stream>>>(query, qbf, (int)(MQ / 8));
    cvt_bf16<<<8192, 256, 0, stream>>>(key,   kbf, (int)(MK / 8));
    cvt_bf16<<<8192, 256, 0, stream>>>(value, vbf, (int)(MK / 8));

    // 2) projections (WMMA bf16, async-to-LDS pipeline)
    gemm_bf16_async<1, 0><<<dim3(4, 64),  256, 0, stream>>>(qbf, wt_q, b_q, qb, nullptr, 8192);
    gemm_bf16_async<1, 0><<<dim3(4, 256), 256, 0, stream>>>(kbf, wt_k, b_k, kb, nullptr, 32768);
    gemm_bf16_async<0, 0><<<dim3(4, 256), 256, 0, stream>>>(vbf, wt_v, b_v, vb, nullptr, 32768);

    // 3) attention downsampling + residual + LN1 (fp32 + bf16 copies)
    attn_ln1<<<8192, 256, 0, stream>>>(qb, kb, vb, ln1_g, ln1_b, o1, o1b);

    // 4) FFN: hidden emitted directly in bf16, second GEMM back to fp32
    gemm_bf16_async<1, 1><<<dim3(4, 64), 256, 0, stream>>>(o1b,  wt_1, fb1, nullptr, hidb, 8192);
    gemm_bf16_async<0, 0><<<dim3(4, 64), 256, 0, stream>>>(hidb, wt_2, fb2, ffnout, nullptr, 8192);

    // 5) residual + LN2 -> output
    add_ln2<<<8192, 256, 0, stream>>>(o1, ffnout, ln2_g, ln2_b, outp);
}
